// MultiExpertMultiHeadAttention_37469294690864
// MI455X (gfx1250) — compile-verified, matrix operand order checked by reading, not run on hardware
//
#include <hip/hip_runtime.h>
#include <hip/hip_bf16.h>
#include <math.h>

// Problem constants (from reference)
#define B_   8
#define S_   512
#define D_   256
#define E_   8
#define H_   4
#define DH_  64
#define NC_  2048            // E * DK (projection output channels)
#define SCALE_ 0.125f        // DH^-0.5

typedef __attribute__((ext_vector_type(16))) __bf16 v16bf;
typedef __attribute__((ext_vector_type(8)))  float  v8f;

// ---- helpers -------------------------------------------------------------

static __device__ __forceinline__ unsigned short f32_to_bf16(float f) {
  unsigned u = __builtin_bit_cast(unsigned, f);
  unsigned r = u + 0x7fffu + ((u >> 16) & 1u);   // round-to-nearest-even
  return (unsigned short)(r >> 16);
}
static __device__ __forceinline__ __bf16 us_to_bf(unsigned short s) {
  return __builtin_bit_cast(__bf16, s);
}

// 16-bit A/B fragment layout (ISA 7.12.2): lane-half 0 holds K {0..7,16..23},
// lane-half 1 holds K {8..15,24..31} -> two contiguous 16B chunks -> b128 x2.
static __device__ __forceinline__ v16bf load_frag_bf16(const unsigned short* base, int half) {
  v16bf f;
  const unsigned short* p0 = base + 8 * half;
  const unsigned short* p1 = base + 16 + 8 * half;
#pragma unroll
  for (int j = 0; j < 8; ++j) { f[j] = us_to_bf(p0[j]); f[8 + j] = us_to_bf(p1[j]); }
  return f;
}
static __device__ __forceinline__ v8f wmma_bf16(v16bf a, v16bf b, v8f c) {
  return __builtin_amdgcn_wmma_f32_16x16x32_bf16(false, a, false, b, (short)0, c,
                                                 false, false);
}

// ---- kernel 0: elementwise f32 -> bf16 (packed, coalesced) ---------------
__global__ __launch_bounds__(256)
void cvt_bf16_kernel(const float* __restrict__ X, unsigned short* __restrict__ Y, int n) {
  int i = (blockIdx.x * 256 + threadIdx.x) * 4;
  if (i < n) {
    float4 v = *(const float4*)(X + i);
    unsigned long long pack =
        (unsigned long long)f32_to_bf16(v.x)
      | ((unsigned long long)f32_to_bf16(v.y) << 16)
      | ((unsigned long long)f32_to_bf16(v.z) << 32)
      | ((unsigned long long)f32_to_bf16(v.w) << 48);
    *(unsigned long long*)(Y + i) = pack;
  }
}

// ---- kernel 1: weight transpose + f32->bf16 ------------------------------
// Wt[n][k] = bf16(W[k][n]);  W is [256][2048] row-major, Wt is [2048][256].
__global__ __launch_bounds__(256)
void transpose_w_kernel(const float* __restrict__ W, unsigned short* __restrict__ Wt) {
  int idx = blockIdx.x * blockDim.x + threadIdx.x;  // total = 2048*256
  int n = idx >> 8;
  int k = idx & 255;
  Wt[idx] = f32_to_bf16(W[(size_t)k * NC_ + n]);
}

// ---- kernel 2: fused QKV projection (WMMA GEMM) --------------------------
// M = B*S = 4096, N = 2048, K = 256.  Each wave computes a 16x128 strip
// (8 accumulators) so every A-fragment feeds 8 WMMAs.
// Q -> [b,e,h,s,dh] (pre-scaled), K -> [b,e,h,s,dh], V -> [b,e,h,dh,s].
__global__ __launch_bounds__(128)
void qkv_proj_kernel(const unsigned short* __restrict__ xq,
                     const unsigned short* __restrict__ xk,
                     const unsigned short* __restrict__ xv,
                     const unsigned short* __restrict__ wq_t,
                     const unsigned short* __restrict__ wk_t,
                     const unsigned short* __restrict__ wv_t,
                     unsigned short* __restrict__ q_ws,
                     unsigned short* __restrict__ k_ws,
                     unsigned short* __restrict__ v_ws) {
  const int lane = threadIdx.x & 31;
  const int wid  = threadIdx.x >> 5;
  const int half = lane >> 4;
  const int mr   = lane & 15;
  // tasks: 3 mats * 256 m-tiles * 16 n-groups = 12288
  int task = blockIdx.x * 4 + wid;
  int mat  = task / 4096;
  int t    = task % 4096;
  int tm   = t >> 4;          // m-tile (16 rows)
  int ng   = t & 15;          // n-group (128 cols)

  const unsigned short* X  = (mat == 0) ? xq : (mat == 1) ? xk : xv;
  const unsigned short* Wt = (mat == 0) ? wq_t : (mat == 1) ? wk_t : wv_t;
  const unsigned short* arow  = X  + (size_t)(tm * 16 + mr) * D_;
  const unsigned short* bbase = Wt + (size_t)(ng * 128 + mr) * D_;

  v8f acc[8] = {};
#pragma unroll
  for (int kb = 0; kb < D_ / 32; ++kb) {
    if (kb < D_ / 32 - 1)
      __builtin_prefetch(arow + (kb + 1) * 32, 0, 0);   // global_prefetch_b8
    v16bf a = load_frag_bf16(arow + kb * 32, half);
#pragma unroll
    for (int n = 0; n < 8; ++n) {
      v16bf b = load_frag_bf16(bbase + (size_t)n * 16 * D_ + kb * 32, half);
      acc[n] = wmma_bf16(a, b, acc[n]);
    }
  }

#pragma unroll
  for (int n = 0; n < 8; ++n) {
    int c  = ng * 128 + n * 16 + mr;       // output channel
    int e  = c >> 8;
    int h  = (c >> 6) & (H_ - 1);
    int dh = c & (DH_ - 1);
#pragma unroll
    for (int r = 0; r < 8; ++r) {
      int row = tm * 16 + r + 8 * half;    // token index in [0, B*S)
      int b_ = row / S_, s = row % S_;
      size_t beh = (size_t)(b_ * E_ + e) * H_ + h;
      float val = acc[n][r];
      if (mat == 0)      q_ws[(beh * S_ + s) * DH_ + dh] = f32_to_bf16(val * SCALE_);
      else if (mat == 1) k_ws[(beh * S_ + s) * DH_ + dh] = f32_to_bf16(val);
      else               v_ws[(beh * DH_ + dh) * S_ + s] = f32_to_bf16(val);
    }
  }
}

// ---- kernel 3: flash attention per (b,e,h), 16 query rows per wave -------
__global__ __launch_bounds__(128)
void attn_kernel(const unsigned short* __restrict__ q_ws,
                 const unsigned short* __restrict__ k_ws,
                 const unsigned short* __restrict__ v_ws,
                 const unsigned char* __restrict__ mask,
                 unsigned short* __restrict__ ctx_ws) {
  __shared__ unsigned short p_lds[4][16 * 32];   // P-tile bounce, 1 KB/wave
  const int lane = threadIdx.x & 31;
  const int wid  = threadIdx.x >> 5;
  const int half = lane >> 4;
  const int mr   = lane & 15;

  int task = blockIdx.x * 4 + wid;      // [0, 256 * 32)
  int qt   = task & 31;                 // query tile (16 rows)
  int beh  = task >> 5;                 // (b*E + e)*H + h
  int h = beh & (H_ - 1);
  int e = (beh >> 2) & (E_ - 1);
  int b = beh >> 5;

  const unsigned short* Qp = q_ws + ((size_t)beh * S_ + qt * 16) * DH_;
  const unsigned short* Kp = k_ws + (size_t)beh * S_ * DH_;
  const unsigned short* Vp = v_ws + (size_t)beh * DH_ * S_;   // [dh][s]
  const unsigned char*  Mp = mask + ((size_t)b * S_ + qt * 16) * S_;

  // A-operand Q fragments (row = mr), dh split into two K=32 chunks
  v16bf qf0 = load_frag_bf16(Qp + mr * DH_, half);
  v16bf qf1 = load_frag_bf16(Qp + mr * DH_ + 32, half);

  v8f o0 = {}, o1 = {}, o2 = {}, o3 = {};
  float m_run[8], l_run[8];
#pragma unroll
  for (int r = 0; r < 8; ++r) { m_run[r] = -INFINITY; l_run[r] = 0.f; }

  for (int kt = 0; kt < S_ / 32; ++kt) {
    int k0 = kt * 32;
    v8f z = {};
    // logits tiles: cols k0..k0+15 and k0+16..k0+31
    v16bf kf = load_frag_bf16(Kp + (size_t)(k0 + mr) * DH_, half);
    v8f s0 = wmma_bf16(qf0, kf, z);
    kf = load_frag_bf16(Kp + (size_t)(k0 + mr) * DH_ + 32, half);
    s0 = wmma_bf16(qf1, kf, s0);
    kf = load_frag_bf16(Kp + (size_t)(k0 + 16 + mr) * DH_, half);
    v8f s1 = wmma_bf16(qf0, kf, z);
    kf = load_frag_bf16(Kp + (size_t)(k0 + 16 + mr) * DH_ + 32, half);
    s1 = wmma_bf16(qf1, kf, s1);

    // mask: logits*(1-m) + (-1e18)*m
#pragma unroll
    for (int r = 0; r < 8; ++r) {
      int qrow = r + 8 * half;
      if (Mp[(size_t)qrow * S_ + k0 + mr])      s0[r] = -1e18f;
      if (Mp[(size_t)qrow * S_ + k0 + 16 + mr]) s1[r] = -1e18f;
    }

    // online softmax (row stats live per C-layout row; reduce over 16 lanes)
#pragma unroll
    for (int r = 0; r < 8; ++r) {
      float tmax = fmaxf(s0[r], s1[r]);
#pragma unroll
      for (int off = 1; off < 16; off <<= 1)
        tmax = fmaxf(tmax, __shfl_xor(tmax, off, 32));
      float mnew = fmaxf(m_run[r], tmax);
      float corr = __expf(m_run[r] - mnew);
      m_run[r] = mnew;
      float p0 = __expf(s0[r] - mnew);
      float p1 = __expf(s1[r] - mnew);
      float rs = p0 + p1;
#pragma unroll
      for (int off = 1; off < 16; off <<= 1)
        rs += __shfl_xor(rs, off, 32);
      l_run[r] = l_run[r] * corr + rs;
      o0[r] *= corr; o1[r] *= corr; o2[r] *= corr; o3[r] *= corr;
      int prow = r + 8 * half;
      p_lds[wid][prow * 32 + mr]      = f32_to_bf16(p0);
      p_lds[wid][prow * 32 + 16 + mr] = f32_to_bf16(p1);
    }

    // Same-wave LDS: DS ops are in-order; compiler inserts s_wait_dscnt.
    // C-layout -> A-layout transpose via LDS, then P(16x32) @ V(32x64).
    v16bf pf = load_frag_bf16(&p_lds[wid][mr * 32], half);
    v16bf vf = load_frag_bf16(Vp + (size_t)(0 * 16 + mr) * S_ + k0, half);
    o0 = wmma_bf16(pf, vf, o0);
    vf = load_frag_bf16(Vp + (size_t)(1 * 16 + mr) * S_ + k0, half);
    o1 = wmma_bf16(pf, vf, o1);
    vf = load_frag_bf16(Vp + (size_t)(2 * 16 + mr) * S_ + k0, half);
    o2 = wmma_bf16(pf, vf, o2);
    vf = load_frag_bf16(Vp + (size_t)(3 * 16 + mr) * S_ + k0, half);
    o3 = wmma_bf16(pf, vf, o3);
  }

  // normalize and write ctx[b][s][e][h*64 + d] as bf16 for the output GEMM
#pragma unroll
  for (int r = 0; r < 8; ++r) {
    int q = qt * 16 + r + 8 * half;
    float inv = 1.f / l_run[r];
    size_t rowbase = ((size_t)(b * S_ + q) * E_ + e) * 256 + h * DH_;
    ctx_ws[rowbase + 0 * 16 + mr] = f32_to_bf16(o0[r] * inv);
    ctx_ws[rowbase + 1 * 16 + mr] = f32_to_bf16(o1[r] * inv);
    ctx_ws[rowbase + 2 * 16 + mr] = f32_to_bf16(o2[r] * inv);
    ctx_ws[rowbase + 3 * 16 + mr] = f32_to_bf16(o3[r] * inv);
  }
}

// ---- kernel 4: output GEMM  ctx[32768,256] @ Wo[256,2048] -> f32 out -----
// Each wave computes a 16x128 strip (8 accumulators, 8x A reuse).
__global__ __launch_bounds__(128)
void out_gemm_kernel(const unsigned short* __restrict__ ctx_ws,
                     const unsigned short* __restrict__ wo_t,
                     float* __restrict__ out) {
  const int lane = threadIdx.x & 31;
  const int wid  = threadIdx.x >> 5;
  const int half = lane >> 4;
  const int mr   = lane & 15;
  // tasks: 2048 m-tiles * 16 n-groups = 32768
  int task = blockIdx.x * 4 + wid;
  int tm = task >> 4;
  int ng = task & 15;

  const unsigned short* arow  = ctx_ws + (size_t)(tm * 16 + mr) * 256;
  const unsigned short* bbase = wo_t  + (size_t)(ng * 128 + mr) * 256;

  v8f acc[8] = {};
#pragma unroll
  for (int kb = 0; kb < 256 / 32; ++kb) {
    if (kb < 256 / 32 - 1)
      __builtin_prefetch(arow + (kb + 1) * 32, 0, 0);   // global_prefetch_b8
    v16bf a = load_frag_bf16(arow + kb * 32, half);
#pragma unroll
    for (int n = 0; n < 8; ++n) {
      v16bf b = load_frag_bf16(bbase + (size_t)n * 16 * 256 + kb * 32, half);
      acc[n] = wmma_bf16(a, b, acc[n]);
    }
  }
#pragma unroll
  for (int n = 0; n < 8; ++n) {
#pragma unroll
    for (int r = 0; r < 8; ++r) {
      size_t row = (size_t)(tm * 16 + r + 8 * half);
      out[row * NC_ + ng * 128 + n * 16 + mr] = acc[n][r];
    }
  }
}

// ---- launcher ------------------------------------------------------------

extern "C" void kernel_launch(void* const* d_in, const int* in_sizes, int n_in,
                              void* d_out, int out_size, void* d_ws, size_t ws_size,
                              hipStream_t stream) {
  const float* queries = (const float*)d_in[0];
  const float* keys    = (const float*)d_in[1];
  const float* values  = (const float*)d_in[2];
  const unsigned char* mask = (const unsigned char*)d_in[3];  // jax bool = 1 byte
  const float* Wq = (const float*)d_in[4];
  const float* Wk = (const float*)d_in[5];
  const float* Wv = (const float*)d_in[6];
  const float* Wo = (const float*)d_in[7];
  float* out = (float*)d_out;
  (void)in_sizes; (void)n_in; (void)out_size; (void)ws_size;

  char* ws = (char*)d_ws;
  size_t off = 0;
  auto alloc = [&](size_t bytes) -> char* {
    char* p = ws + off;
    off += (bytes + 255) & ~(size_t)255;
    return p;
  };
  unsigned short* wq_t  = (unsigned short*)alloc((size_t)NC_ * D_ * 2);      // 1 MB each
  unsigned short* wk_t  = (unsigned short*)alloc((size_t)NC_ * D_ * 2);
  unsigned short* wv_t  = (unsigned short*)alloc((size_t)NC_ * D_ * 2);
  unsigned short* wo_t  = (unsigned short*)alloc((size_t)NC_ * D_ * 2);
  unsigned short* xq_bf = (unsigned short*)alloc((size_t)B_ * S_ * D_ * 2);  // 2 MB each
  unsigned short* xk_bf = (unsigned short*)alloc((size_t)B_ * S_ * D_ * 2);
  unsigned short* xv_bf = (unsigned short*)alloc((size_t)B_ * S_ * D_ * 2);
  unsigned short* q_ws  = (unsigned short*)alloc((size_t)B_ * E_ * H_ * S_ * DH_ * 2); // 16.8 MB
  unsigned short* k_ws  = (unsigned short*)alloc((size_t)B_ * E_ * H_ * S_ * DH_ * 2);
  unsigned short* v_ws  = (unsigned short*)alloc((size_t)B_ * E_ * H_ * S_ * DH_ * 2);
  unsigned short* ctx_ws= (unsigned short*)alloc((size_t)B_ * S_ * E_ * 256 * 2);

  // 0) activations f32 -> bf16 (one pass, so GEMM loops are conversion-free)
  {
    int n = B_ * S_ * D_;                 // 1,048,576
    int blocks = (n / 4 + 255) / 256;
    cvt_bf16_kernel<<<blocks, 256, 0, stream>>>(queries, xq_bf, n);
    cvt_bf16_kernel<<<blocks, 256, 0, stream>>>(keys,    xk_bf, n);
    cvt_bf16_kernel<<<blocks, 256, 0, stream>>>(values,  xv_bf, n);
  }

  // 1) weight transpose + bf16 convert
  dim3 tb(256), tg((NC_ * D_) / 256);
  transpose_w_kernel<<<tg, tb, 0, stream>>>(Wq, wq_t);
  transpose_w_kernel<<<tg, tb, 0, stream>>>(Wk, wk_t);
  transpose_w_kernel<<<tg, tb, 0, stream>>>(Wv, wv_t);
  transpose_w_kernel<<<tg, tb, 0, stream>>>(Wo, wo_t);

  // 2) QKV projection: 3 * 256 * 16 = 12288 wave strips, 4 waves/block
  qkv_proj_kernel<<<12288 / 4, 128, 0, stream>>>(xq_bf, xk_bf, xv_bf,
                                                 wq_t, wk_t, wv_t,
                                                 q_ws, k_ws, v_ws);

  // 3) attention: B*E*H * (S/16) = 8192 wave tasks
  attn_kernel<<<8192 / 4, 128, 0, stream>>>(q_ws, k_ws, v_ws, mask, ctx_ws);

  // 4) output GEMM: 2048 * 16 = 32768 wave strips
  out_gemm_kernel<<<32768 / 4, 128, 0, stream>>>(ctx_ws, wo_t, out);
}
